// SparseCrossLayerAttention_76227079569754
// MI455X (gfx1250) — matrix operand order repaired and checked by reading, not compiled, not run on hardware
//
#include <hip/hip_runtime.h>
#include <math.h>

typedef __attribute__((ext_vector_type(16))) _Float16 v16h;
typedef __attribute__((ext_vector_type(8)))  float    v8f;
typedef __attribute__((ext_vector_type(4)))  unsigned v4u;
typedef __attribute__((ext_vector_type(8)))  int      v8i;
typedef __attribute__((ext_vector_type(4)))  int      v4i;

#define NPOS 4096   // 64*64
#define HGT  64
#define WID  64

#if defined(__has_builtin)
#if __has_builtin(__builtin_amdgcn_tensor_load_to_lds)
#define HAS_TDM 1
#endif
#endif

#if defined(HAS_TDM)
#if __has_include(<hip/amd_detail/amd_gfx1250_TDM.h>)
#define TDM_6ARG 1
#endif
#endif

#if defined(__has_builtin)
#if __has_builtin(__builtin_amdgcn_s_wait_tensorcnt)
#define WAIT_TENSOR(n) __builtin_amdgcn_s_wait_tensorcnt(n)
#else
#define WAIT_TENSOR(n) asm volatile("s_wait_tensorcnt %0" ::"i"(n))
#endif
#else
#define WAIT_TENSOR(n) asm volatile("s_wait_tensorcnt %0" ::"i"(n))
#endif

// ---------------------------------------------------------------------------
// f32 -> f16 elementwise convert (input feature + weights)
// ---------------------------------------------------------------------------
__global__ void cvt_f32_f16_kernel(const float* __restrict__ src,
                                   _Float16* __restrict__ dst, int n) {
    int i = blockIdx.x * 256 + threadIdx.x;
    if (i < n) dst[i] = (_Float16)src[i];
}

// ---------------------------------------------------------------------------
// 2x bilinear upsample (half-pixel), [B,256,32,32] f32 -> [B,256,64,64] f16
// ---------------------------------------------------------------------------
__global__ void upsample2x_kernel(const float* __restrict__ hi,
                                  _Float16* __restrict__ out, int total) {
    int idx = blockIdx.x * blockDim.x + threadIdx.x;
    if (idx >= total) return;
    int x  = idx & 63;
    int y  = (idx >> 6) & 63;
    int bc = idx >> 12;                        // b*256 + c
    const float* src = hi + (size_t)bc * 1024; // 32*32
    float fx = 0.5f * (float)x - 0.25f;
    float fy = 0.5f * (float)y - 0.25f;
    float x0f = floorf(fx), y0f = floorf(fy);
    float wx = fx - x0f, wy = fy - y0f;
    int x0 = min(max((int)x0f, 0), 31);
    int x1 = min(max((int)x0f + 1, 0), 31);
    int y0 = min(max((int)y0f, 0), 31);
    int y1 = min(max((int)y0f + 1, 0), 31);
    float v00 = src[y0 * 32 + x0], v01 = src[y0 * 32 + x1];
    float v10 = src[y1 * 32 + x0], v11 = src[y1 * 32 + x1];
    out[idx] = (_Float16)((1.f - wx) * (1.f - wy) * v00 + wx * (1.f - wy) * v01 +
                          (1.f - wx) * wy * v10 + wx * wy * v11);
}

// ---------------------------------------------------------------------------
// TDM: issue a 2D f16 tile load into LDS (tile_d0 elems/row x tile_d1 rows,
// row stride in elements).  LDS pad: 1 dword after each 2^pad_code * 8 bytes
// (code 3 -> every 64B, code 5 -> every 256B).
// ---------------------------------------------------------------------------
#if defined(HAS_TDM)
__device__ __forceinline__ void tdm_load_2d_f16(unsigned lds_off,
                                                const _Float16* gptr,
                                                unsigned tile_d0, unsigned tile_d1,
                                                unsigned long long row_stride_elems,
                                                unsigned pad_code) {
    unsigned long long ga = (unsigned long long)(size_t)gptr;
    v4u g0;
    g0[0] = 1u;                                   // count=1 valid user descriptor
    g0[1] = lds_off;                              // lds_addr (bytes)
    g0[2] = (unsigned)(ga & 0xffffffffull);       // global_addr[31:0]
    g0[3] = (unsigned)((ga >> 32) & 0x01ffffffull) | (2u << 30); // addr[56:32]+type=2
    unsigned td0 = 1u << 20, td1 = 1u << 20;      // generous tensor dims (no OOB clip)
    v8i g1;
    g1[0] = (int)((1u << 16)                      // data_size = 2 bytes
                  | (1u << 20)                    // pad_enable
                  | (pad_code << 22)              // pad_interval
                  | (0u << 25));                  // pad_amount = 1 dword
    g1[1] = (int)((td0 & 0xffffu) << 16);         // tensor_dim0[15:0]
    g1[2] = (int)(((td0 >> 16) & 0xffffu) | ((td1 & 0xffffu) << 16));
    g1[3] = (int)(((td1 >> 16) & 0xffffu) | ((tile_d0 & 0xffffu) << 16));
    g1[4] = (int)(tile_d1 & 0xffffu);             // tile_dim1; tile_dim2 = 0
    g1[5] = (int)(unsigned)(row_stride_elems & 0xffffffffull);
    g1[6] = (int)(unsigned)((row_stride_elems >> 32) & 0xffffull);
    g1[7] = 0;
    v4i g2 = {0, 0, 0, 0};
    v4i g3 = {0, 0, 0, 0};
#if defined(TDM_6ARG)
    v8i g4 = {0, 0, 0, 0, 0, 0, 0, 0};
    __builtin_amdgcn_tensor_load_to_lds(g0, g1, g2, g3, g4, 0);
#else
    __builtin_amdgcn_tensor_load_to_lds(g0, g1, g2, g3, 0);
#endif
}
#endif

// ---------------------------------------------------------------------------
// Fragment loaders from padded LDS tiles.
// A tile rows: 68B stride (32 f16 + 4B pad).
// B tile rows: 520B stride (256 f16 + 4B pad after each 256B half).
// ---------------------------------------------------------------------------
__device__ __forceinline__ v16h load_afrag(const unsigned char* Abuf, int lane) {
    const int arow = lane & 15;
    const int kba  = (lane < 16) ? 0 : 8;
    v16h a;
    #pragma unroll
    for (int i = 0; i < 8; ++i)
        a[i] = *(const _Float16*)(Abuf + arow * 68 + (kba + i) * 2);
    #pragma unroll
    for (int i = 0; i < 8; ++i)
        a[8 + i] = *(const _Float16*)(Abuf + arow * 68 + (kba + 16 + i) * 2);
    return a;
}

__device__ __forceinline__ v16h load_bfrag(const unsigned char* Bbuf, int n, int kbb) {
    const unsigned off_n = (unsigned)(n * 2 + ((n >> 7) << 2)); // +4B past half-row
    v16h bf;
    #pragma unroll
    for (int i = 0; i < 16; ++i)
        bf[i] = *(const _Float16*)(Bbuf + (kbb + (i < 8 ? i : i + 8)) * 520 + off_n);
    return bf;
}

// One K-step: wave tile 16(O) x 32(N) -> A reused across two WMMAs
__device__ __forceinline__ void wmma_step(const unsigned char* Abuf,
                                          const unsigned char* Bbuf,
                                          int lane, int wv,
                                          v8f& acc0, v8f& acc1) {
    const int kbb = (lane < 16) ? 0 : 8;
    const int nb  = wv * 32 + (lane & 15);
    v16h a  = load_afrag(Abuf, lane);
    v16h b0 = load_bfrag(Bbuf, nb, kbb);
    v16h b1 = load_bfrag(Bbuf, nb + 16, kbb);
    acc0 = __builtin_amdgcn_wmma_f32_16x16x32_f16(false, a, false, b0,
                                                  (short)0, acc0, false, false);
    acc1 = __builtin_amdgcn_wmma_f32_16x16x32_f16(false, a, false, b1,
                                                  (short)0, acc1, false, false);
}

// ---------------------------------------------------------------------------
// Generic 1x1-conv GEMM (f16 in, f32 accumulate via WMMA):
//   Y[b,o,n] = epilogue( sum_c W[o,c] * X[b,c,n] )
// Channel-split input at Csplit (fused concat).  Block = 256 thr = 8 waves;
// block tile = 16(O) x 256(N); wave tile 16x32 (2 WMMA / K-step).
// Grid: (Npos/256, O/16, B).  Staging: TDM double-buffered (wave 0 drives
// the DMA; tail peeled for clean s_wait_tensorcnt codegen), manual fallback.
// ---------------------------------------------------------------------------
__global__ void gemm1x1_wmma(const _Float16* __restrict__ X1, long xs1,
                             const _Float16* __restrict__ X2, long xs2,
                             int Csplit, int C, int Npos,
                             const _Float16* __restrict__ Wt, int O,
                             const float* __restrict__ bias,
                             float* __restrict__ Yf32,
                             _Float16* __restrict__ Yf16, long ys,
                             const float* __restrict__ gamma,
                             const float* __restrict__ beta,
                             const float* __restrict__ rmean,
                             const float* __restrict__ rvar) {
    __shared__ __align__(16) unsigned char ldsA[2][1088];   // 16 rows * 68B
    __shared__ __align__(16) unsigned char ldsB[2][16640];  // 32 rows * 520B

    const int tid  = threadIdx.x;
    const int lane = tid & 31;
    const int wv   = tid >> 5;
    const int n0   = blockIdx.x * 256;
    const int o0   = blockIdx.y * 16;
    const int b    = blockIdx.z;

    const _Float16* X1b = X1 + (size_t)b * xs1;
    const _Float16* X2b = X2 + (size_t)b * xs2;
    const int T = C / 32;

    v8f acc0 = {};
    v8f acc1 = {};

#if defined(HAS_TDM)
    if (wv == 0) {  // wave-uniform branch: only wave 0 issues TDM
        tdm_load_2d_f16((unsigned)(size_t)&ldsA[0][0],
                        Wt + (size_t)o0 * C, 32, 16, (unsigned long long)C, 3);
        tdm_load_2d_f16((unsigned)(size_t)&ldsB[0][0],
                        X1b + n0, 256, 32, (unsigned long long)Npos, 5);
    }
    int t = 0;
    for (; t < T - 1; ++t) {   // steady state: prefetch t+1, wait for t
        if (wv == 0) {
            const int k0 = (t + 1) * 32;
            const _Float16* src = (k0 < Csplit)
                                      ? (X1b + (size_t)k0 * Npos)
                                      : (X2b + (size_t)(k0 - Csplit) * Npos);
            const int nb = (t + 1) & 1;
            tdm_load_2d_f16((unsigned)(size_t)&ldsA[nb][0],
                            Wt + (size_t)o0 * C + k0, 32, 16,
                            (unsigned long long)C, 3);
            tdm_load_2d_f16((unsigned)(size_t)&ldsB[nb][0],
                            src + n0, 256, 32, (unsigned long long)Npos, 5);
            WAIT_TENSOR(2);    // in-order TDM: tile t's 2 ops complete
        }
        __syncthreads();
        wmma_step(ldsA[t & 1], ldsB[t & 1], lane, wv, acc0, acc1);
        __syncthreads();
    }
    if (wv == 0) WAIT_TENSOR(0);   // tail tile
    __syncthreads();
    wmma_step(ldsA[t & 1], ldsB[t & 1], lane, wv, acc0, acc1);
#else
    for (int t = 0; t < T; ++t) {
        const int k0 = t * 32;
        const _Float16* src = (k0 < Csplit)
                                  ? (X1b + (size_t)k0 * Npos)
                                  : (X2b + (size_t)(k0 - Csplit) * Npos);
        {   // B tile: thread -> row k = tid>>3, 32 contiguous n
            const int kk = tid >> 3;
            const int nb = (tid & 7) * 32;
            const _Float16* gs = src + (size_t)kk * Npos + n0 + nb;
            unsigned char* row = ldsB[0] + kk * 520;
            #pragma unroll
            for (int i = 0; i < 32; ++i) {
                int n = nb + i;
                *(_Float16*)(row + n * 2 + ((n >> 7) << 2)) = gs[i];
            }
        }
        if (tid < 128) {  // A tile: o = tid>>3, 4 contiguous k
            const int o  = tid >> 3;
            const int kk = (tid & 7) * 4;
            const _Float16* gs = Wt + (size_t)(o0 + o) * C + k0 + kk;
            unsigned char* drow = ldsA[0] + o * 68 + kk * 2;
            #pragma unroll
            for (int i = 0; i < 4; ++i) *(_Float16*)(drow + i * 2) = gs[i];
        }
        __syncthreads();
        wmma_step(ldsA[0], ldsB[0], lane, wv, acc0, acc1);
        __syncthreads();
    }
#endif

    // C/D layout: VGPR r -> M = r + 8*(lane>=16), N = lane&15
    const int mbase = (lane < 16) ? 0 : 8;
    const int nlo   = n0 + wv * 32 + (lane & 15);
    #pragma unroll
    for (int r = 0; r < 8; ++r) {
        const int o = o0 + mbase + r;
        float s = 1.f, t2 = 0.f;
        float bia = bias ? bias[o] : 0.f;
        if (gamma) {
            s  = gamma[o] * rsqrtf(rvar[o] + 1e-5f);
            t2 = beta[o] - rmean[o] * s;
        }
        float v0 = acc0[r] + bia;
        float v1 = acc1[r] + bia;
        if (gamma) {
            v0 = fmaxf(v0 * s + t2, 0.f);
            v1 = fmaxf(v1 * s + t2, 0.f);
        }
        size_t base = (size_t)b * ys + (size_t)o * Npos;
        if (Yf32) { Yf32[base + nlo] = v0; Yf32[base + nlo + 16] = v1; }
        if (Yf16) { Yf16[base + nlo] = (_Float16)v0; Yf16[base + nlo + 16] = (_Float16)v1; }
    }
}

// ---------------------------------------------------------------------------
// Deformable point-sampling attention.
// One wave per query: lane = sample point (bilinear k-sample -> logit),
// LDS softmax over 32 points, then 4 v-channels/lane weighted accumulation.
// Block = 256 thr = 8 queries.  Grid = B*4096/8.
// ---------------------------------------------------------------------------
__global__ void deform_attn_kernel(const float* __restrict__ q,   // [B,16,N]
                                   const float* __restrict__ k,   // [B,16,N]
                                   const float* __restrict__ v,   // [B,128,N]
                                   const float* __restrict__ off, // [B,64,N]
                                   _Float16* __restrict__ out) {  // [B,128,N] f16
    __shared__ int   sIdx[8][32][4];
    __shared__ float sWgt[8][32][4];
    __shared__ float sLog[8][32];
    __shared__ float sExp[8][32];

    const int lane = threadIdx.x & 31;
    const int wv   = threadIdx.x >> 5;
    const int qidx = blockIdx.x * 8 + wv;
    const int b = qidx >> 12;
    const int n = qidx & 4095;
    const int x = n & 63, y = n >> 6;

    const float* qb = q   + (size_t)b * 16  * NPOS;
    const float* kb = k   + (size_t)b * 16  * NPOS;
    const float* vb = v   + (size_t)b * 128 * NPOS;
    const float* ob = off + (size_t)b * 64  * NPOS;

    // ---- phase 1: per-lane sample point -> logit
    const int p = lane;
    float dx = ob[(size_t)(2 * p)     * NPOS + n];
    float dy = ob[(size_t)(2 * p + 1) * NPOS + n];
    float sx = (float)x + dx, sy = (float)y + dy;
    float x0f = floorf(sx), y0f = floorf(sy);
    float wx = sx - x0f, wy = sy - y0f;
    int x0 = min(max((int)x0f, 0), WID - 1);
    int x1 = min(max((int)x0f + 1, 0), WID - 1);
    int y0 = min(max((int)y0f, 0), HGT - 1);
    int y1 = min(max((int)y0f + 1, 0), HGT - 1);
    int i00 = y0 * WID + x0, i01 = y0 * WID + x1;
    int i10 = y1 * WID + x0, i11 = y1 * WID + x1;
    float w00 = (1.f - wx) * (1.f - wy), w01 = wx * (1.f - wy);
    float w10 = (1.f - wx) * wy,         w11 = wx * wy;

    sIdx[wv][p][0] = i00; sIdx[wv][p][1] = i01;
    sIdx[wv][p][2] = i10; sIdx[wv][p][3] = i11;
    sWgt[wv][p][0] = w00; sWgt[wv][p][1] = w01;
    sWgt[wv][p][2] = w10; sWgt[wv][p][3] = w11;

    float logit = 0.f;
    #pragma unroll
    for (int c = 0; c < 16; ++c) {
        const float* kc = kb + (size_t)c * NPOS;
        float ks = w00 * kc[i00] + w01 * kc[i01] + w10 * kc[i10] + w11 * kc[i11];
        logit += qb[(size_t)c * NPOS + n] * ks;
    }
    logit *= 0.25f;                    // 1/sqrt(INTER_C=16)
    sLog[wv][p] = logit;
    __syncthreads();

    // ---- softmax over the 32 points
    float mx = -INFINITY;
    #pragma unroll
    for (int i = 0; i < 32; ++i) mx = fmaxf(mx, sLog[wv][i]);
    sExp[wv][p] = __expf(logit - mx);
    __syncthreads();
    float denom = 0.f;
    #pragma unroll
    for (int i = 0; i < 32; ++i) denom += sExp[wv][i];
    float inv = 1.f / denom;

    // ---- phase 2: 4 v-channels per lane, sum over 32 points
    float acc[4] = {0.f, 0.f, 0.f, 0.f};
    for (int pp = 0; pp < 32; ++pp) {
        float a  = sExp[wv][pp] * inv;
        int   j00 = sIdx[wv][pp][0], j01 = sIdx[wv][pp][1];
        int   j10 = sIdx[wv][pp][2], j11 = sIdx[wv][pp][3];
        float u00 = sWgt[wv][pp][0], u01 = sWgt[wv][pp][1];
        float u10 = sWgt[wv][pp][2], u11 = sWgt[wv][pp][3];
        #pragma unroll
        for (int j = 0; j < 4; ++j) {
            const float* vc = vb + (size_t)(lane + 32 * j) * NPOS;
            float vs = u00 * vc[j00] + u01 * vc[j01] + u10 * vc[j10] + u11 * vc[j11];
            acc[j] += a * vs;
        }
    }
    _Float16* outb = out + (size_t)b * 128 * NPOS;
    #pragma unroll
    for (int j = 0; j < 4; ++j)
        outb[(size_t)(lane + 32 * j) * NPOS + n] = (_Float16)acc[j];
}

// ---------------------------------------------------------------------------
extern "C" void kernel_launch(void* const* d_in, const int* in_sizes, int n_in,
                              void* d_out, int out_size, void* d_ws, size_t ws_size,
                              hipStream_t stream) {
    const float* low   = (const float*)d_in[0];   // [2,256,64,64]
    const float* high  = (const float*)d_in[1];   // [2,256,32,32]
    const float* W1    = (const float*)d_in[2];   // [128,512]
    const float* b1    = (const float*)d_in[3];
    const float* W2    = (const float*)d_in[4];   // [128,256]
    const float* b2    = (const float*)d_in[5];
    const float* Wq    = (const float*)d_in[6];   // [16,128]
    const float* bq    = (const float*)d_in[7];
    const float* Wk    = (const float*)d_in[8];   // [16,128]
    const float* bk    = (const float*)d_in[9];
    const float* Wv    = (const float*)d_in[10];  // [128,128]
    const float* bv    = (const float*)d_in[11];
    const float* Woff  = (const float*)d_in[12];  // [64,128]
    const float* boff  = (const float*)d_in[13];
    const float* Wb    = (const float*)d_in[14];  // [256,384]
    const float* gamma = (const float*)d_in[15];
    const float* beta  = (const float*)d_in[16];
    const float* rmean = (const float*)d_in[17];
    const float* rvar  = (const float*)d_in[18];
    float* out = (float*)d_out;

    const size_t NC = (size_t)2 * 256 * NPOS;   // 2,097,152
    const size_t NI = (size_t)2 * 128 * NPOS;   // 1,048,576
    const size_t NQ = (size_t)2 * 16  * NPOS;   //   131,072
    const size_t NO = (size_t)2 * 64  * NPOS;   //   524,288

    // workspace layout
    char* p = (char*)d_ws;
    _Float16* high16 = (_Float16*)p; p += NC * 2;
    _Float16* low16  = (_Float16*)p; p += NC * 2;
    _Float16* qf16   = (_Float16*)p; p += NI * 2;
    _Float16* vf16   = (_Float16*)p; p += NI * 2;
    _Float16* attn16 = (_Float16*)p; p += NI * 2;
    float*    qbuf   = (float*)p;    p += NQ * 4;
    float*    kbuf   = (float*)p;    p += NQ * 4;
    float*    vbuf   = (float*)p;    p += NI * 4;
    float*    offb   = (float*)p;    p += NO * 4;
    _Float16* W1h    = (_Float16*)p; p += (size_t)128 * 512 * 2;
    _Float16* W2h    = (_Float16*)p; p += (size_t)128 * 256 * 2;
    _Float16* Wqh    = (_Float16*)p; p += (size_t)16  * 128 * 2;
    _Float16* Wkh    = (_Float16*)p; p += (size_t)16  * 128 * 2;
    _Float16* Wvh    = (_Float16*)p; p += (size_t)128 * 128 * 2;
    _Float16* Woh    = (_Float16*)p; p += (size_t)64  * 128 * 2;
    _Float16* Wbh    = (_Float16*)p; p += (size_t)256 * 384 * 2;

    const long sCH = (long)256 * NPOS;
    const long sIN = (long)128 * NPOS;
    const long sIC = (long)16  * NPOS;
    const long sOF = (long)64  * NPOS;

    // 0) one-time f16 conversions
    cvt_f32_f16_kernel<<<(int)((NC + 255) / 256), 256, 0, stream>>>(low, low16, (int)NC);
    cvt_f32_f16_kernel<<<(128 * 512 + 255) / 256, 256, 0, stream>>>(W1, W1h, 128 * 512);
    cvt_f32_f16_kernel<<<(128 * 256 + 255) / 256, 256, 0, stream>>>(W2, W2h, 128 * 256);
    cvt_f32_f16_kernel<<<(16 * 128 + 255) / 256, 256, 0, stream>>>(Wq, Wqh, 16 * 128);
    cvt_f32_f16_kernel<<<(16 * 128 + 255) / 256, 256, 0, stream>>>(Wk, Wkh, 16 * 128);
    cvt_f32_f16_kernel<<<(128 * 128 + 255) / 256, 256, 0, stream>>>(Wv, Wvh, 128 * 128);
    cvt_f32_f16_kernel<<<(64 * 128 + 255) / 256, 256, 0, stream>>>(Woff, Woh, 64 * 128);
    cvt_f32_f16_kernel<<<(256 * 384 + 255) / 256, 256, 0, stream>>>(Wb, Wbh, 256 * 384);

    // 1) bilinear 2x upsample of high feature -> f16
    upsample2x_kernel<<<(int)(NC / 256), 256, 0, stream>>>(high, high16, (int)NC);

    // 2) query_feature = W1 * concat(high_up, low) + b1   (O=128, C=512) -> f16
    gemm1x1_wmma<<<dim3(16, 8, 2), 256, 0, stream>>>(
        high16, sCH, low16, sCH, 256, 512, NPOS, W1h, 128, b1,
        nullptr, qf16, sIN, nullptr, nullptr, nullptr, nullptr);

    // 3) value_feature = W2 * high_up + b2                (O=128, C=256) -> f16
    gemm1x1_wmma<<<dim3(16, 8, 2), 256, 0, stream>>>(
        high16, sCH, high16, sCH, 256, 256, NPOS, W2h, 128, b2,
        nullptr, vf16, sIN, nullptr, nullptr, nullptr, nullptr);

    // 4) q / k / v / offset projections -> f32 (consumed by gather kernel)
    gemm1x1_wmma<<<dim3(16, 1, 2), 256, 0, stream>>>(
        qf16, sIN, qf16, sIN, 128, 128, NPOS, Wqh, 16, bq,
        qbuf, nullptr, sIC, nullptr, nullptr, nullptr, nullptr);
    gemm1x1_wmma<<<dim3(16, 1, 2), 256, 0, stream>>>(
        vf16, sIN, vf16, sIN, 128, 128, NPOS, Wkh, 16, bk,
        kbuf, nullptr, sIC, nullptr, nullptr, nullptr, nullptr);
    gemm1x1_wmma<<<dim3(16, 8, 2), 256, 0, stream>>>(
        vf16, sIN, vf16, sIN, 128, 128, NPOS, Wvh, 128, bv,
        vbuf, nullptr, sIN, nullptr, nullptr, nullptr, nullptr);
    gemm1x1_wmma<<<dim3(16, 4, 2), 256, 0, stream>>>(
        qf16, sIN, qf16, sIN, 128, 128, NPOS, Woh, 64, boff,
        offb, nullptr, sOF, nullptr, nullptr, nullptr, nullptr);

    // 5) deformable point-sampling attention -> f16
    deform_attn_kernel<<<1024, 256, 0, stream>>>(qbuf, kbuf, vbuf, offb, attn16);

    // 6) out = relu(BN(Wb * concat(attn_out, high_up)))   (O=256, C=384) -> f32
    gemm1x1_wmma<<<dim3(16, 16, 2), 256, 0, stream>>>(
        attn16, sIN, high16, sCH, 128, 384, NPOS, Wbh, 256, nullptr,
        out, nullptr, sCH, gamma, beta, rmean, rvar);
}
